// MultiHeadSelfAttention_26843545600593
// MI455X (gfx1250) — matrix operand order repaired and checked by reading, not compiled, not run on hardware
//
#include <hip/hip_runtime.h>
#include <hip/hip_bf16.h>

typedef __attribute__((ext_vector_type(16))) _Float16 v16h;
typedef __attribute__((ext_vector_type(8)))  _Float16 v8h;
typedef __attribute__((ext_vector_type(8)))  float    v8f;
typedef __attribute__((ext_vector_type(4)))  unsigned int v4u;
typedef __attribute__((ext_vector_type(8)))  unsigned int v8u;

#define WMMA_F16(a,b,c) __builtin_amdgcn_wmma_f32_16x16x32_f16(false,(a),false,(b),(short)0,(c),false,false)

constexpr int Bb = 2, Ss = 4096, Dd = 512, Hh = 8, Dh = 64;
constexpr int M_ROWS = Bb * Ss;          // 8192
constexpr int LSTR = 40;                 // padded LDS row stride (f16): 64B data + 16B pad

__device__ __forceinline__ v8h ld8(const _Float16* p) { return *(const v8h*)p; }
__device__ __forceinline__ v16h cat16(v8h a, v8h b) {
  return __builtin_shufflevector(a, b, 0,1,2,3,4,5,6,7,8,9,10,11,12,13,14,15);
}

// ---------------------------------------------------------------- conversions
__global__ void cvt_f32_f16(const float* __restrict__ src, _Float16* __restrict__ dst, int n) {
  int i = blockIdx.x * blockDim.x + threadIdx.x;
  int stride = gridDim.x * blockDim.x;
  for (; i < n; i += stride) dst[i] = (_Float16)src[i];
}

// 512x512 transpose + convert: dst[n][k] = (f16)src[k][n]
__global__ void cvtT_f32_f16(const float* __restrict__ src, _Float16* __restrict__ dst) {
  const int i = blockIdx.x * blockDim.x + threadIdx.x;   // 0..262143
  const int n = i >> 9, k = i & 511;
  dst[i] = (_Float16)src[k * 512 + n];
}

// --------------------------------------------------- TDM tile load (B tile)
// DMA a 2D tile: 64 rows (n) x 32 cols (k) of f16, global row stride 512
// elements, into LDS with 16B padding after every 64B row (matches LSTR=40).
__device__ __forceinline__ void tdm_load_64x32(const _Float16* g, unsigned int lds_addr) {
  const unsigned long long ga = (unsigned long long)g;
  v4u g0;
  g0[0] = 1u;                                      // count=1, user descriptor
  g0[1] = lds_addr;                                // LDS byte address
  g0[2] = (unsigned int)ga;                        // global_addr[31:0]
  g0[3] = (unsigned int)(ga >> 32) | 0x80000000u;  // global_addr[56:32] | type=2
  v8u g1;
  g1[0] = (1u << 16)        // data_size = 2 bytes
        | (1u << 20)        // pad_enable
        | (3u << 22)        // pad_interval: 16 DWORDs (64B) between pads
        | (3u << 25);       // pad_amount: 4 DWORDs (16B)
  g1[1] = 32u << 16;        // tensor_dim0 = 32 (k extent)
  g1[2] = 64u << 16;        // tensor_dim1 = 64 (n extent)
  g1[3] = 32u << 16;        // tile_dim0 = 32
  g1[4] = 64u;              // tile_dim1 = 64, tile_dim2 = 0
  g1[5] = 512u;             // tensor_dim0_stride = 512 elements
  g1[6] = 0u;
  g1[7] = 0u;
  asm volatile("tensor_load_to_lds %0, %1" :: "s"(g0), "s"(g1) : "memory");
}

// ------------------------------------------------------ GEMM tile staging
// A tile 64x32 (contiguous rows): async global->LDS b128, 2 issues/thread.
// B tile 64x32 from pre-transposed weights: one TDM issue from wave 0.
__device__ __forceinline__ void stage_tiles(const _Float16* __restrict__ A,
                                            const _Float16* __restrict__ BT,
                                            int kbase, int tid,
                                            _Float16 (*As)[LSTR],
                                            _Float16 (*Bt)[LSTR]) {
  #pragma unroll
  for (int i = 0; i < 2; ++i) {
    const int idx = i * 128 + tid;            // 0..255 -> 8 f16 (16B) each
    const int r  = idx >> 2;                  // 0..63
    const int cc = (idx & 3) * 8;             // 0,8,16,24
    const unsigned int lds = (unsigned int)(unsigned long long)&As[r][cc];
    const unsigned long long g =
        (unsigned long long)(A + (size_t)r * Dd + kbase + cc);
    asm volatile("global_load_async_to_lds_b128 %0, %1, off"
                 :: "v"(lds), "v"(g) : "memory");
  }
  if (tid < 32)
    tdm_load_64x32(BT + kbase, (unsigned int)(unsigned long long)&Bt[0][0]);
}

// -------------------------------------------- double-buffered GEMM mainloop
// Block computes 64x64; wave w owns rows w*16..w*16+15 and all 64 cols
// (4 accumulators -> 4 WMMA per staged K-step of 32). Tile k+1 staging
// (async + TDM) fully overlapped with tile k compute.
__device__ __forceinline__ void gemm_main(const _Float16* __restrict__ A,
                                          const _Float16* __restrict__ BT,
                                          int tid,
                                          _Float16 (*As)[64][LSTR],
                                          _Float16 (*Bt)[64][LSTR],
                                          v8f c[4]) {
  const int wid = tid >> 5, lane = tid & 31;
  const int r16 = lane & 15;
  const int kb2 = (lane < 16) ? 0 : 16;
  const int kbB = (lane < 16) ? 0 : 16;
  stage_tiles(A, BT, 0, tid, As[0], Bt[0]);
  int buf = 0;
  for (int kt = 0; kt < Dd / 32; ++kt) {
    __builtin_amdgcn_s_wait_tensorcnt(0);
    asm volatile("s_wait_asynccnt 0" ::: "memory");
    __syncthreads();
    if (kt + 1 < Dd / 32)
      stage_tiles(A, BT, (kt + 1) * 32, tid, As[buf ^ 1], Bt[buf ^ 1]);
    v16h a = cat16(ld8(&As[buf][wid * 16 + r16][kb2]),
                   ld8(&As[buf][wid * 16 + r16][kb2 + 8]));
    #pragma unroll
    for (int t = 0; t < 4; ++t) {
      v16h bf = cat16(ld8(&Bt[buf][t * 16 + r16][kbB]),
                      ld8(&Bt[buf][t * 16 + r16][kbB + 8]));
      c[t] = WMMA_F16(a, bf, c[t]);
    }
    buf ^= 1;
  }
}

// ------------------------------------------------------------- QKV GEMM
// C = xh[8192,512] * W{q,k,v}^T-fed + b. Q scaled by 1/sqrt(Dh).
// Q,K scattered to [B,H,S,Dh]; V scattered TRANSPOSED to [B,H,Dh,S].
__global__ __launch_bounds__(128) void qkv_gemm(
    const _Float16* __restrict__ xh, const _Float16* __restrict__ wqkvT,
    const float* __restrict__ bq, const float* __restrict__ bk,
    const float* __restrict__ bv,
    _Float16* __restrict__ Qd, _Float16* __restrict__ Kd, _Float16* __restrict__ Vd) {
  __shared__ __align__(16) _Float16 As[2][64][LSTR];
  __shared__ __align__(16) _Float16 Bt[2][64][LSTR];
  const int tid = threadIdx.x;
  const int m0 = blockIdx.x * 64;
  const int nt = blockIdx.y;           // 0..23
  const int mat = nt >> 3;             // 0=Q 1=K 2=V
  const int n0 = (nt & 7) * 64;
  v8f z = {};
  v8f c[4] = {z, z, z, z};
  gemm_main(xh + (size_t)m0 * Dd,
            wqkvT + (size_t)mat * Dd * 512 + (size_t)n0 * Dd,   // [N][K] rows
            tid, As, Bt, c);

  const int wid = tid >> 5, lane = tid & 31;
  const int r16 = lane & 15;
  const int b = m0 / Ss;
  const int sbase = (m0 % Ss) + wid * 16 + ((lane < 16) ? 0 : 8);
  const int h = n0 >> 6;               // 64-col block is h-aligned
  const float* bias = (mat == 0) ? bq : (mat == 1 ? bk : bv);
  const float scale = (mat == 0) ? 0.125f : 1.0f;
  #pragma unroll
  for (int t = 0; t < 4; ++t) {
    const int f = t * 16 + r16;
    const float bsv = bias[n0 + f];
    if (mat < 2) {
      _Float16* dst = ((mat == 0) ? Qd : Kd)
                      + (size_t)((b * Hh + h) * Ss) * Dh + f;
      #pragma unroll
      for (int v = 0; v < 8; ++v)
        dst[(size_t)(sbase + v) * Dh] = (_Float16)((c[t][v] + bsv) * scale);
    } else {
      _Float16* dst = Vd + ((size_t)(b * Hh + h) * Dh + f) * Ss + sbase;
      #pragma unroll
      for (int v = 0; v < 8; ++v)       // consecutive v -> consecutive s
        dst[v] = (_Float16)(c[t][v] + bsv);
    }
  }
}

// --------------------------------------------------------- flash attention
// One block: 64 query rows for one (b,h); 4 waves x 16 rows; 32-key blocks.
// K^T and V (pre-transposed [Dh][S]) fragments load straight from global;
// only wave-private LDS (P round-trip) -> no barriers at all.
__global__ __launch_bounds__(128) void flash_attn(
    const _Float16* __restrict__ Qd, const _Float16* __restrict__ Kd,
    const _Float16* __restrict__ Vd, _Float16* __restrict__ Ao) {
  __shared__ __align__(16) _Float16 Pw[4][16][LSTR];
  const int tid = threadIdx.x, wid = tid >> 5, lane = tid & 31;
  const int bh = blockIdx.y;                // b*H + h
  const int b = bh / Hh, h = bh % Hh;
  const int q0 = blockIdx.x * 64;
  const _Float16* Qp = Qd + (size_t)bh * Ss * Dh;   // [S][Dh]
  const _Float16* Kp = Kd + (size_t)bh * Ss * Dh;   // [S][Dh]
  const _Float16* Vp = Vd + (size_t)bh * Dh * Ss;   // [Dh][S]

  const int r16 = lane & 15;
  const int kb2 = (lane < 16) ? 0 : 16;
  const int kbB = (lane < 16) ? 0 : 16;

  // Q fragments for this wave's 16 rows (contraction over Dh=64 -> 2 frags)
  const _Float16* qrow = Qp + (size_t)(q0 + wid * 16 + r16) * Dh;
  const v16h qf0 = cat16(ld8(qrow + kb2),      ld8(qrow + kb2 + 8));
  const v16h qf1 = cat16(ld8(qrow + 32 + kb2), ld8(qrow + 32 + kb2 + 8));

  v8f z = {};
  v8f o[4] = {z, z, z, z};
  float mrun[8], lrun[8];
  #pragma unroll
  for (int v = 0; v < 8; ++v) { mrun[v] = -1e30f; lrun[v] = 0.0f; }

  for (int k0 = 0; k0 < Ss; k0 += 32) {
    // S = Q x K^T (two 16-key subtiles)
    v8f s0 = z, s1 = z;
    {
      const _Float16* kr0 = Kp + (size_t)(k0 + r16) * Dh;
      const _Float16* kr1 = kr0 + 16 * Dh;
      v16h ka = cat16(ld8(kr0 + kbB),      ld8(kr0 + kbB + 8));
      v16h kb = cat16(ld8(kr0 + 32 + kbB), ld8(kr0 + 32 + kbB + 8));
      s0 = WMMA_F16(qf0, ka, s0);
      s0 = WMMA_F16(qf1, kb, s0);
      ka = cat16(ld8(kr1 + kbB),      ld8(kr1 + kbB + 8));
      kb = cat16(ld8(kr1 + 32 + kbB), ld8(kr1 + 32 + kbB + 8));
      s1 = WMMA_F16(qf0, ka, s1);
      s1 = WMMA_F16(qf1, kb, s1);
    }

    // online softmax (rows per-v, spread over 16-lane halves)
    float p0[8], p1[8];
    #pragma unroll
    for (int v = 0; v < 8; ++v) {
      float mt = fmaxf(s0[v], s1[v]);
      #pragma unroll
      for (int off = 1; off < 16; off <<= 1)
        mt = fmaxf(mt, __shfl_xor(mt, off, 32));
      const float mn = fmaxf(mrun[v], mt);
      const float alpha = __expf(mrun[v] - mn);
      p0[v] = __expf(s0[v] - mn);
      p1[v] = __expf(s1[v] - mn);
      float rs = p0[v] + p1[v];
      #pragma unroll
      for (int off = 1; off < 16; off <<= 1)
        rs += __shfl_xor(rs, off, 32);
      lrun[v] = lrun[v] * alpha + rs;
      mrun[v] = mn;
      #pragma unroll
      for (int t = 0; t < 4; ++t) o[t][v] *= alpha;
    }

    // P: C-layout f32 -> wave-private LDS -> A-layout f16 fragment
    {
      const int prow = (lane < 16) ? 0 : 8;
      #pragma unroll
      for (int v = 0; v < 8; ++v) {
        Pw[wid][prow + v][r16]      = (_Float16)p0[v];
        Pw[wid][prow + v][16 + r16] = (_Float16)p1[v];
      }
    }
    const v16h pf = cat16(ld8(&Pw[wid][r16][kb2]), ld8(&Pw[wid][r16][kb2 + 8]));

    // O += P x V, V fragments straight from global (contraction contiguous)
    #pragma unroll
    for (int t = 0; t < 4; ++t) {
      const _Float16* vr = Vp + (size_t)(t * 16 + r16) * Ss + k0 + kbB;
      v16h vf = cat16(ld8(vr), ld8(vr + 8));
      o[t] = WMMA_F16(pf, vf, o[t]);
    }
  }

  // writeback to Ao[B*S][H*Dh] (GEMM-ready for output projection)
  const int rowb = q0 + wid * 16 + ((lane < 16) ? 0 : 8);
  const int colb = h * 64 + r16;
  #pragma unroll
  for (int v = 0; v < 8; ++v) {
    const float inv = 1.0f / lrun[v];
    _Float16* d = Ao + ((size_t)(b * Ss + rowb + v)) * 512 + colb;
    #pragma unroll
    for (int t = 0; t < 4; ++t)
      d[t * 16] = (_Float16)(o[t][v] * inv);
  }
}

// ------------------------------------------------------- output projection
__global__ __launch_bounds__(128) void out_proj(
    const _Float16* __restrict__ Ao, const _Float16* __restrict__ woT,
    const float* __restrict__ bo, float* __restrict__ out) {
  __shared__ __align__(16) _Float16 As[2][64][LSTR];
  __shared__ __align__(16) _Float16 Bt[2][64][LSTR];
  const int tid = threadIdx.x;
  const int m0 = blockIdx.x * 64;
  const int n0 = blockIdx.y * 64;
  v8f z = {};
  v8f c[4] = {z, z, z, z};
  gemm_main(Ao + (size_t)m0 * Dd, woT + (size_t)n0 * Dd, tid, As, Bt, c);

  const int wid = tid >> 5, lane = tid & 31;
  const int rbase = m0 + wid * 16 + ((lane < 16) ? 0 : 8);
  #pragma unroll
  for (int t = 0; t < 4; ++t) {
    const int col = n0 + t * 16 + (lane & 15);
    const float bsv = bo[col];
    #pragma unroll
    for (int v = 0; v < 8; ++v)
      out[(size_t)(rbase + v) * Dd + col] = c[t][v] + bsv;
  }
}

// --------------------------------------------------------------- launcher
extern "C" void kernel_launch(void* const* d_in, const int* in_sizes, int n_in,
                              void* d_out, int out_size, void* d_ws, size_t ws_size,
                              hipStream_t stream) {
  (void)in_sizes; (void)n_in; (void)out_size; (void)ws_size;
  const float* x  = (const float*)d_in[0];
  const float* Wq = (const float*)d_in[1];
  const float* bq = (const float*)d_in[2];
  const float* Wk = (const float*)d_in[3];
  const float* bk = (const float*)d_in[4];
  const float* Wv = (const float*)d_in[5];
  const float* bv = (const float*)d_in[6];
  const float* Wo = (const float*)d_in[7];
  const float* bo = (const float*)d_in[8];
  float* out = (float*)d_out;

  char* ws = (char*)d_ws;
  _Float16* xh    = (_Float16*)(ws + 0);          //  8192*512 f16 =  8 MiB
  _Float16* wqkvT = (_Float16*)(ws +  8388608);   //  3*512*512 transposed
  _Float16* woT   = (_Float16*)(ws +  9961472);   //  512*512 transposed
  _Float16* Qd    = (_Float16*)(ws + 10485760);   //  [B,H,S,Dh], pre-scaled
  _Float16* Kd    = (_Float16*)(ws + 18874368);   //  [B,H,S,Dh]
  _Float16* Vd    = (_Float16*)(ws + 27262976);   //  [B,H,Dh,S] (transposed)
  _Float16* Ao    = (_Float16*)(ws + 35651584);   //  [B*S, H*Dh]

  cvt_f32_f16<<<dim3(2048), dim3(256), 0, stream>>>(x, xh, M_ROWS * Dd);
  cvtT_f32_f16<<<dim3(1024), dim3(256), 0, stream>>>(Wq, wqkvT);
  cvtT_f32_f16<<<dim3(1024), dim3(256), 0, stream>>>(Wk, wqkvT + 262144);
  cvtT_f32_f16<<<dim3(1024), dim3(256), 0, stream>>>(Wv, wqkvT + 524288);
  cvtT_f32_f16<<<dim3(1024), dim3(256), 0, stream>>>(Wo, woT);

  qkv_gemm<<<dim3(M_ROWS / 64, 24), dim3(128), 0, stream>>>(
      xh, wqkvT, bq, bk, bv, Qd, Kd, Vd);

  flash_attn<<<dim3(Ss / 64, Bb * Hh), dim3(128), 0, stream>>>(Qd, Kd, Vd, Ao);

  out_proj<<<dim3(M_ROWS / 64, Dd / 64), dim3(128), 0, stream>>>(Ao, woT, bo, out);
}
